// OrthoLinear_24885040513145
// MI455X (gfx1250) — matrix operand order repaired and checked by reading, not compiled, not run on hardware
//
#include <hip/hip_runtime.h>

typedef __attribute__((ext_vector_type(16))) _Float16 v16h;
typedef __attribute__((ext_vector_type(8)))  float    v8f;
typedef __attribute__((ext_vector_type(2)))  _Float16 h2;

#define IN_F  4096
#define OUT_F 4096
#define BM 128
#define BN 64
#define BK 32

// ---------------------------------------------------------------------------
// Kernel 1: dequantize int4 base weights -> dense f16 W [OUT_F][IN_F] (row-major)
// packed[r][c2] holds two nibbles: low -> col 2*c2, high -> col 2*c2+1
// ---------------------------------------------------------------------------
__global__ void dequant_kernel(const int* __restrict__ packed,
                               const float* __restrict__ scales,
                               _Float16* __restrict__ W) {
  const int halfIn = IN_F / 2;
  int idx = blockIdx.x * blockDim.x + threadIdx.x;
  if (idx >= OUT_F * halfIn) return;
  int row = idx / halfIn;
  int c2  = idx - row * halfIn;
  int p   = packed[idx];
  float s = scales[row];
  h2 o;
  o.x = (_Float16)((float)((p & 15) - 8) * s);
  o.y = (_Float16)((float)(((p >> 4) & 15) - 8) * s);
  *(h2*)(W + (size_t)row * IN_F + 2 * c2) = o;
}

// ---------------------------------------------------------------------------
// Kernel 2: scatter-add CSR residual (uniform nnz/row) into f16 W
// Each (row,col) appears exactly once -> 16-bit RMW is race-free.
// ---------------------------------------------------------------------------
__global__ void residual_kernel(const _Float16* __restrict__ vals,
                                const int* __restrict__ cols,
                                int nnz, int per_row,
                                _Float16* __restrict__ W) {
  int i = blockIdx.x * blockDim.x + threadIdx.x;
  if (i >= nnz) return;
  int row = i / per_row;
  size_t o = (size_t)row * IN_F + cols[i];
  W[o] = (_Float16)((float)W[o] + (float)vals[i]);   // alpha = 1.0
}

// ---------------------------------------------------------------------------
// Kernel 3: out[M][OUT_F] f32 = X[M][IN_F] (f32 -> f16) * W[OUT_F][IN_F]^T
// 128 threads = 4 wave32. Block tile 128(M) x 64(N), K-step 32 (one WMMA K).
// Each wave owns a 32(M) x 64(N) register tile: 8 accumulators; every B
// fragment is reused by two A fragments -> 8 wmma per K-step per wave.
// ---------------------------------------------------------------------------
__global__ void __launch_bounds__(128)
gemm_kernel(const float* __restrict__ X, const _Float16* __restrict__ W,
            float* __restrict__ out) {
  __shared__ _Float16 Xs[BM][BK];   // [m_local][k_local]
  __shared__ _Float16 Wt[BK][BN];   // [k_local][n_local]  (transposed stage)

  const int tid  = threadIdx.x;
  const int lane = tid & 31;
  const int wv   = tid >> 5;                 // 0..3
  const int m0   = blockIdx.y * BM;
  const int n0   = blockIdx.x * BN;

  v8f acc[8] = {};                           // [strip s][ntile t] -> acc[s*4+t]

  // cooperative-load coordinates (128 threads)
  const float*    xp = X + (size_t)(m0 + tid) * IN_F;      // one 32-col row each
  const int wn  = tid >> 1;                  // W tile n 0..63
  const int wk0 = (tid & 1) * 16;            // W tile k 0 or 16
  const _Float16* wp = W + (size_t)(n0 + wn) * IN_F + wk0;

  for (int k0 = 0; k0 < IN_F; k0 += BK) {
    // stage X tile: f32 global -> f16 LDS (row-major), 32 floats per thread
    #pragma unroll
    for (int j = 0; j < BK; j += 2) {
      float2 f = *(const float2*)(xp + k0 + j);
      h2 hh; hh.x = (_Float16)f.x; hh.y = (_Float16)f.y;
      *(h2*)&Xs[tid][j] = hh;
    }
    // stage W tile transposed: Wt[k][n] = W[n0+n][k0+k]
    #pragma unroll
    for (int j = 0; j < 16; ++j) Wt[wk0 + j][wn] = wp[k0 + j];

    if (k0 + BK < IN_F) {                    // prefetch next K tile
      __builtin_prefetch(xp + k0 + BK, 0, 0);
      __builtin_prefetch(wp + k0 + BK, 0, 0);
    }
    __syncthreads();

    // A fragments (16x32 f16, M x K), two M-strips per wave:
    // lanes 0-15 -> K {0-7,16-23}, lanes 16-31 -> K {8-15,24-31}
    union { v16h v; unsigned u[8]; } A[2];
    const int kb = (lane < 16) ? 0 : 8;
    #pragma unroll
    for (int s = 0; s < 2; ++s) {
      const int ml = wv * 32 + s * 16 + (lane & 15);
      #pragma unroll
      for (int v = 0; v < 4; ++v) {
        A[s].u[v]     = *(const unsigned*)&Xs[ml][kb + 2 * v];
        A[s].u[4 + v] = *(const unsigned*)&Xs[ml][16 + kb + 2 * v];
      }
    }

    #pragma unroll
    for (int t = 0; t < 4; ++t) {
      // B fragment (32x16 f16, K x N): N = lane&15; same K order as A
      union { v16h v; unsigned u[8]; } B;
      #pragma unroll
      for (int v = 0; v < 8; ++v) {
        B.v[v]     = Wt[kb + v][t * 16 + (lane & 15)];
        B.v[8 + v] = Wt[16 + kb + v][t * 16 + (lane & 15)];
      }
      acc[t]     = __builtin_amdgcn_wmma_f32_16x16x32_f16(
          false, A[0].v, false, B.v, (short)0, acc[t],     false, false);
      acc[4 + t] = __builtin_amdgcn_wmma_f32_16x16x32_f16(
          false, A[1].v, false, B.v, (short)0, acc[4 + t], false, false);
    }
    __syncthreads();
  }

  // C/D layout: VGPR r -> M = r (lanes 0-15) or r+8 (lanes 16-31); N = lane&15
  const int ncol = n0 + (lane & 15);
  #pragma unroll
  for (int s = 0; s < 2; ++s) {
    const int mbase = m0 + wv * 32 + s * 16 + ((lane >> 4) << 3);
    #pragma unroll
    for (int t = 0; t < 4; ++t)
      #pragma unroll
      for (int r = 0; r < 8; ++r)
        out[(size_t)(mbase + r) * OUT_F + ncol + t * 16] = acc[s * 4 + t][r];
  }
}

// ---------------------------------------------------------------------------
extern "C" void kernel_launch(void* const* d_in, const int* in_sizes, int n_in,
                              void* d_out, int out_size, void* d_ws, size_t ws_size,
                              hipStream_t stream) {
  const float*    x           = (const float*)d_in[0];
  const int*      base_packed = (const int*)d_in[1];
  const float*    scales      = (const float*)d_in[2];
  const _Float16* ortho_vals  = (const _Float16*)d_in[3];
  const int*      ortho_cols  = (const int*)d_in[4];
  // d_in[5] = ortho_ptr: uniform arange*nnz_per_row by construction; row = i / per_row
  float*    out = (float*)d_out;
  _Float16* W   = (_Float16*)d_ws;           // 4096*4096*2 = 32 MB scratch

  const int M       = in_sizes[0] / IN_F;    // 8192 tokens
  const int nnz     = in_sizes[3];
  const int per_row = nnz / OUT_F;

  const int packedN = OUT_F * (IN_F / 2);
  dequant_kernel<<<(packedN + 255) / 256, 256, 0, stream>>>(base_packed, scales, W);
  residual_kernel<<<(nnz + 255) / 256, 256, 0, stream>>>(ortho_vals, ortho_cols,
                                                         nnz, per_row, W);
  dim3 grid(OUT_F / BN, M / BM);
  gemm_kernel<<<grid, 128, 0, stream>>>(x, W, out);
}
